// LSTMAE_56049323213457
// MI455X (gfx1250) — compile-verified, matrix operand order checked by reading, not run on hardware
//
#include <hip/hip_runtime.h>
#include <hip/hip_bf16.h>

typedef __attribute__((ext_vector_type(16))) __bf16 v16bf;
typedef __attribute__((ext_vector_type(8)))  __bf16 v8bf;
typedef __attribute__((ext_vector_type(8)))  float  v8f;

#define HDIM   128
#define TSTEPS 8

#define WMMA_BF16(A, B, C) \
    __builtin_amdgcn_wmma_f32_16x16x32_bf16(false, (A), false, (B), (short)0, (C), false, false)

// ---- fragment helpers -------------------------------------------------------
// 16-bit A-matrix 16x32 layout (ISA 7.12.2): lanes 0-15 hold row M=lane,
// K=0..7 (VGPR0-3) and K=16..23 (VGPR4-7); lanes 16-31 hold K=8..15 / 24..31.
// B (NxK mirrored: lane = N column, same K packing) loaded from f32 weights.

__device__ __forceinline__ v16bf load_afrag_bf16(const __bf16* __restrict__ row,
                                                 int kt, int lane) {
    const __bf16* p = row + kt * 32 + ((lane & 16) ? 8 : 0);
    v8bf lo = *(const v8bf*)p;
    v8bf hi = *(const v8bf*)(p + 16);
    return __builtin_shufflevector(lo, hi, 0,1,2,3,4,5,6,7,8,9,10,11,12,13,14,15);
}

template<int ROWLEN>
__device__ __forceinline__ v16bf load_wfrag_f32(const float* __restrict__ rowptr,
                                                int kt, int lane) {
    v16bf f;
    const int kb = kt * 32 + ((lane & 16) ? 8 : 0);
#pragma unroll
    for (int j = 0; j < 8; ++j) {
        const int k0 = kb + j, k1 = kb + 16 + j;
        float a = (k0 < ROWLEN) ? rowptr[k0] : 0.f;
        float b = (k1 < ROWLEN) ? rowptr[k1] : 0.f;
        f[j]     = (__bf16)a;
        f[j + 8] = (__bf16)b;
    }
    return f;
}

__device__ __forceinline__ v16bf zero_frag() {
    v16bf f;
#pragma unroll
    for (int j = 0; j < 16; ++j) f[j] = (__bf16)0.f;
    return f;
}

__device__ __forceinline__ v8f bcast8(float x) {
    v8f v = {x, x, x, x, x, x, x, x};
    return v;
}

// Branchless transcendentals: one v_exp_f32 each, no libm range-reduction
// branches (keeps EXEC untouched inside the timestep loop, TRANS ops
// co-execute with the WMMA pipe).
__device__ __forceinline__ float sigm(float x) { return 1.f / (1.f + __expf(-x)); }
__device__ __forceinline__ float tnh(float x)  {
    // tanh(x) = 1 - 2/(exp(2x)+1); saturates exactly at +/-1 for large |x|
    float e = __expf(2.f * x);
    return 1.f - 2.f / (e + 1.f);
}

// ---- fused LSTM layer kernel ------------------------------------------------
// Block = 256 threads = 8 waves; batch tile = 16 rows. Wave w owns gate
// n-tiles {w, 8+w, 16+w, 24+w} (cols w*16..w*16+15 of each of i/f/g/o), so
// the c/h elementwise update is wave-local. h(t) is exchanged through LDS.

template<int DIN, int DINP, bool IN_F32, bool CONST_IN,
         bool WRITE_SEQ, bool WRITE_Z, bool WRITE_OUT>
__global__ __launch_bounds__(256) void lstm_stage(
    const float*  __restrict__ xf,      // enc0 input (f32, [B,T,DIN])
    const __bf16* __restrict__ xb,      // bf16 input ([B,T,DINP] or [B,DINP])
    const float*  __restrict__ Wih, const float* __restrict__ Whh,
    const float*  __restrict__ bih, const float* __restrict__ bhh,
    const float*  __restrict__ Wex, const float* __restrict__ bex, // W_lat/W_out
    __bf16* __restrict__ seq_out,       // [B,T,128] bf16
    __bf16* __restrict__ z_out,         // [B,64]    bf16
    float*  __restrict__ out)           // [B,T,60]  f32
{
    constexpr int KIN = DINP / 32;
    const int tid  = threadIdx.x;
    const int w    = tid >> 5;          // wave id 0..7
    const int lane = tid & 31;
    const int nlo  = lane & 15;
    const int mhi  = (lane >> 4) << 3;  // 0 or 8: which row-half this lane holds in C
    const long b0  = (long)blockIdx.x * 16;

    __shared__ __bf16 sH[16][HDIM];                        // h(t) tile, bf16
    __shared__ __bf16 sIn[IN_F32 ? 16 * TSTEPS * 64 : 16]; // enc0 staged input

    // zero h(0)
    for (int i = tid; i < 16 * HDIM / 2; i += 256)
        ((unsigned*)&sH[0][0])[i] = 0u;

    if constexpr (IN_F32) {
        // stage x tile -> bf16, zero-pad DIN..DINP-1
        for (int i = tid; i < 16 * TSTEPS * 64; i += 256) {
            int m = i >> 9;
            int t = (i >> 6) & 7;
            int k = i & 63;
            float v = (k < DIN) ? xf[((b0 + m) * TSTEPS + t) * (long)DIN + k] : 0.f;
            sIn[i] = (__bf16)v;
        }
    }

    // ---- weight fragments in registers (converted f32 -> bf16 once) ----
    v16bf bWhh[4][4];
#pragma unroll
    for (int q = 0; q < 4; ++q)
#pragma unroll
        for (int kt = 0; kt < 4; ++kt)
            bWhh[q][kt] = load_wfrag_f32<HDIM>(
                Whh + (size_t)(q * HDIM + w * 16 + nlo) * HDIM, kt, lane);

    v16bf bWih[4][KIN];
#pragma unroll
    for (int q = 0; q < 4; ++q)
#pragma unroll
        for (int kt = 0; kt < KIN; ++kt)
            bWih[q][kt] = load_wfrag_f32<DIN>(
                Wih + (size_t)(q * HDIM + w * 16 + nlo) * DIN, kt, lane);

    float bias[4];
#pragma unroll
    for (int q = 0; q < 4; ++q)
        bias[q] = bih[q * HDIM + w * 16 + nlo] + bhh[q * HDIM + w * 16 + nlo];

    // extra projection (latent or output head), handled by waves 0..3
    v16bf bEx[4];
    float biasEx = 0.f;
    if constexpr (WRITE_Z || WRITE_OUT) {
        constexpr int EXROWS = WRITE_Z ? 64 : 60;
        if (w < 4) {
            const int row = w * 16 + nlo;
#pragma unroll
            for (int kt = 0; kt < 4; ++kt)
                bEx[kt] = (row < EXROWS)
                              ? load_wfrag_f32<HDIM>(Wex + (size_t)row * HDIM, kt, lane)
                              : zero_frag();
            biasEx = (row < EXROWS) ? bex[row] : 0.f;
        }
    }

    __syncthreads();

    // dec0: input contribution is identical for every timestep -> hoist it
    v8f xg[4];
    if constexpr (CONST_IN) {
#pragma unroll
        for (int q = 0; q < 4; ++q) xg[q] = bcast8(bias[q]);
#pragma unroll
        for (int kt = 0; kt < KIN; ++kt) {
            v16bf a = load_afrag_bf16(xb + (b0 + nlo) * (long)DINP, kt, lane);
#pragma unroll
            for (int q = 0; q < 4; ++q) xg[q] = WMMA_BF16(a, bWih[q][kt], xg[q]);
        }
    }

    v8f c = bcast8(0.f);

    for (int t = 0; t < TSTEPS; ++t) {
        v8f acc[4];
        if constexpr (CONST_IN) {
#pragma unroll
            for (int q = 0; q < 4; ++q) acc[q] = xg[q];
        } else {
#pragma unroll
            for (int q = 0; q < 4; ++q) acc[q] = bcast8(bias[q]);
#pragma unroll
            for (int kt = 0; kt < KIN; ++kt) {
                v16bf a;
                if constexpr (IN_F32)
                    a = load_afrag_bf16(&sIn[(nlo * TSTEPS + t) * 64], kt, lane);
                else
                    a = load_afrag_bf16(xb + ((b0 + nlo) * TSTEPS + t) * (long)DINP,
                                        kt, lane);
#pragma unroll
                for (int q = 0; q < 4; ++q) acc[q] = WMMA_BF16(a, bWih[q][kt], acc[q]);
            }
        }

        // recurrent part: g += h(t-1) @ Whh^T
#pragma unroll
        for (int kt = 0; kt < 4; ++kt) {
            v16bf a = load_afrag_bf16(&sH[nlo][0], kt, lane);
#pragma unroll
            for (int q = 0; q < 4; ++q) acc[q] = WMMA_BF16(a, bWhh[q][kt], acc[q]);
        }

        // gates + state update (PyTorch order i,f,g,o), wave-local, branchless
        v8f hv;
#pragma unroll
        for (int j = 0; j < 8; ++j) {
            float iv = sigm(acc[0][j]);
            float fv = sigm(acc[1][j]);
            float gv = tnh(acc[2][j]);
            float ov = sigm(acc[3][j]);
            float cv = fv * c[j] + iv * gv;
            c[j] = cv;
            hv[j] = ov * tnh(cv);
        }

        __syncthreads();   // everyone done reading h(t-1)
#pragma unroll
        for (int r = 0; r < 8; ++r) {
            __bf16 hb = (__bf16)hv[r];
            sH[mhi + r][w * 16 + nlo] = hb;
            if constexpr (WRITE_SEQ)
                seq_out[((b0 + mhi + r) * TSTEPS + t) * (long)HDIM + w * 16 + nlo] = hb;
        }
        __syncthreads();   // h(t) visible to all waves

        if constexpr (WRITE_OUT) {
            if (w < 4) {   // wave-uniform branch: EXEC all-ones inside
                v8f oacc = bcast8(biasEx);
#pragma unroll
                for (int kt = 0; kt < 4; ++kt) {
                    v16bf a = load_afrag_bf16(&sH[nlo][0], kt, lane);
                    oacc = WMMA_BF16(a, bEx[kt], oacc);
                }
                const int col = w * 16 + nlo;
                if (col < 60) {
#pragma unroll
                    for (int r = 0; r < 8; ++r)
                        out[((b0 + mhi + r) * TSTEPS + t) * 60L + col] = oacc[r];
                }
            }
        }
    }

    if constexpr (WRITE_Z) {
        if (w < 4) {       // z = h_last @ W_lat^T + b_lat
            v8f oacc = bcast8(biasEx);
#pragma unroll
            for (int kt = 0; kt < 4; ++kt) {
                v16bf a = load_afrag_bf16(&sH[nlo][0], kt, lane);
                oacc = WMMA_BF16(a, bEx[kt], oacc);
            }
#pragma unroll
            for (int r = 0; r < 8; ++r)
                z_out[(b0 + mhi + r) * 64L + w * 16 + nlo] = (__bf16)oacc[r];
        }
    }
}

// ---- host side --------------------------------------------------------------

extern "C" void kernel_launch(void* const* d_in, const int* in_sizes, int n_in,
                              void* d_out, int out_size, void* d_ws, size_t ws_size,
                              hipStream_t stream) {
    const float* x       = (const float*)d_in[0];
    const float* e0_Wih  = (const float*)d_in[1];
    const float* e0_Whh  = (const float*)d_in[2];
    const float* e0_bih  = (const float*)d_in[3];
    const float* e0_bhh  = (const float*)d_in[4];
    const float* e1_Wih  = (const float*)d_in[5];
    const float* e1_Whh  = (const float*)d_in[6];
    const float* e1_bih  = (const float*)d_in[7];
    const float* e1_bhh  = (const float*)d_in[8];
    const float* d0_Wih  = (const float*)d_in[9];
    const float* d0_Whh  = (const float*)d_in[10];
    const float* d0_bih  = (const float*)d_in[11];
    const float* d0_bhh  = (const float*)d_in[12];
    const float* d1_Wih  = (const float*)d_in[13];
    const float* d1_Whh  = (const float*)d_in[14];
    const float* d1_bih  = (const float*)d_in[15];
    const float* d1_bhh  = (const float*)d_in[16];
    const float* W_lat   = (const float*)d_in[17];
    const float* b_lat   = (const float*)d_in[18];
    const float* W_out   = (const float*)d_in[19];
    const float* b_out   = (const float*)d_in[20];

    const int B = in_sizes[0] / (TSTEPS * 60);    // 65536

    // workspace layout: seq bf16 [B,T,128] | z bf16 [B,64]
    __bf16* seq0 = (__bf16*)d_ws;
    __bf16* zbuf = (__bf16*)((char*)d_ws + (size_t)B * TSTEPS * HDIM * sizeof(__bf16));
    float*  outp = (float*)d_out;

    dim3 grid(B / 16), block(256);

    // encoder layer 0: x(f32,D=60) -> seq0
    lstm_stage<60, 64, true, false, true, false, false><<<grid, block, 0, stream>>>(
        x, nullptr, e0_Wih, e0_Whh, e0_bih, e0_bhh,
        nullptr, nullptr, seq0, nullptr, nullptr);

    // encoder layer 1: seq0 -> (latent z fused) zbuf
    lstm_stage<128, 128, false, false, false, true, false><<<grid, block, 0, stream>>>(
        nullptr, seq0, e1_Wih, e1_Whh, e1_bih, e1_bhh,
        W_lat, b_lat, nullptr, zbuf, nullptr);

    // decoder layer 0: z broadcast over T -> seq0 (input contribution hoisted)
    lstm_stage<64, 64, false, true, true, false, false><<<grid, block, 0, stream>>>(
        nullptr, zbuf, d0_Wih, d0_Whh, d0_bih, d0_bhh,
        nullptr, nullptr, seq0, nullptr, nullptr);

    // decoder layer 1 + fused output projection -> d_out
    lstm_stage<128, 128, false, false, false, false, true><<<grid, block, 0, stream>>>(
        nullptr, seq0, d1_Wih, d1_Whh, d1_bih, d1_bhh,
        W_out, b_out, nullptr, nullptr, outp);
}
